// MuMoE_7490422964550
// MI455X (gfx1250) — matrix operand order repaired and checked by reading, not compiled, not run on hardware
//
#include <hip/hip_runtime.h>
#include <math.h>

// ---------- CDNA5 WMMA types ----------
typedef __attribute__((ext_vector_type(16))) __bf16 v16bf;
typedef __attribute__((ext_vector_type(8)))  float  v8f;
typedef __attribute__((ext_vector_type(4)))  int    v4i;

union Frag { uint4 u[2]; v16bf v; };
static_assert(sizeof(Frag) == 32, "frag size");

__device__ __forceinline__ unsigned short f2bf(float f) {
  unsigned u = __float_as_uint(f);
  unsigned r = u + 0x7FFFu + ((u >> 16) & 1u);   // round-to-nearest-even
  return (unsigned short)(r >> 16);
}

__device__ __forceinline__ float gelu_exact(float v) {
  return 0.5f * v * (1.0f + erff(v * 0.70710678118654752f));
}

// ---------- CDNA5 async global->LDS path (guarded; falls back to reg staging) ----------
#if __has_builtin(__builtin_amdgcn_global_load_async_to_lds_b128) && \
    __has_builtin(__builtin_amdgcn_s_wait_asynccnt)
#define USE_ASYNC_LDS 1
typedef __attribute__((address_space(1))) v4i as1_v4i;
typedef __attribute__((address_space(3))) v4i as3_v4i;
__device__ __forceinline__ void async_b128(const void* g, const void* l) {
  // flat global VA == AS1 VA; low 32 bits of generic LDS pointer == AS3 offset
  __builtin_amdgcn_global_load_async_to_lds_b128(
      (as1_v4i*)(unsigned long long)(uintptr_t)g,
      (as3_v4i*)(unsigned)(uintptr_t)l, 0, 0);
}
#else
#define USE_ASYNC_LDS 0
#endif

// ---------- converts ----------
__global__ __launch_bounds__(256) void cvt_bf16_kernel(
    const float* __restrict__ in, unsigned short* __restrict__ out, size_t n8) {
  size_t i = (size_t)blockIdx.x * blockDim.x + threadIdx.x;
  if (i >= n8) return;
  const float4* p = (const float4*)(in + i * 8);
  float4 a = p[0], b = p[1];
  uint4 o;
  o.x = (unsigned)f2bf(a.x) | ((unsigned)f2bf(a.y) << 16);
  o.y = (unsigned)f2bf(a.z) | ((unsigned)f2bf(a.w) << 16);
  o.z = (unsigned)f2bf(b.x) | ((unsigned)f2bf(b.y) << 16);
  o.w = (unsigned)f2bf(b.z) | ((unsigned)f2bf(b.w) << 16);
  ((uint4*)out)[i] = o;
}

// out[r*Din + d] = (bf16) in[d*R + r]   (drops bias row; R = 512)
__global__ __launch_bounds__(256) void cvt_T_kernel(
    const float* __restrict__ in, unsigned short* __restrict__ out,
    int Din, size_t total) {
  size_t i = (size_t)blockIdx.x * blockDim.x + threadIdx.x;
  if (i >= total) return;
  int r = (int)(i / (size_t)Din);
  int d = (int)(i % (size_t)Din);
  out[i] = f2bf(in[(size_t)d * 512 + r]);
}

// ---------- gating: proj -> LN -> entmax1.5 -> g1,g2 (fp32, one wave/token) ----------
__global__ __launch_bounds__(256) void gate_kernel(
    const float* __restrict__ x, const float* __restrict__ proj,
    const float* __restrict__ C1, const float* __restrict__ C2,
    float* __restrict__ g1, float* __restrict__ g2) {
  __shared__ float zsort[8][16];
  const int wave = threadIdx.x >> 5, lane = threadIdx.x & 31;
  const int t = blockIdx.x * 8 + wave;
  const float* xr = x + (size_t)t * 1024;

  float acc[16];
#pragma unroll
  for (int e = 0; e < 16; ++e) acc[e] = 0.f;

  for (int d = lane; d < 1024; d += 32) {
    float xv = xr[d];
    const float4* p = (const float4*)(proj + d * 16);
#pragma unroll
    for (int q = 0; q < 4; ++q) {
      float4 pv = p[q];
      acc[4 * q + 0] += xv * pv.x;
      acc[4 * q + 1] += xv * pv.y;
      acc[4 * q + 2] += xv * pv.z;
      acc[4 * q + 3] += xv * pv.w;
    }
  }
#pragma unroll
  for (int e = 0; e < 16; ++e) {
    float v = acc[e];
    v += __shfl_xor(v, 16, 32);
    v += __shfl_xor(v, 8, 32);
    v += __shfl_xor(v, 4, 32);
    v += __shfl_xor(v, 2, 32);
    v += __shfl_xor(v, 1, 32);
    acc[e] = v;   // replicated full sum
  }
  // LayerNorm (no affine) over 16
  float mean = 0.f;
#pragma unroll
  for (int e = 0; e < 16; ++e) mean += acc[e];
  mean *= 0.0625f;
  float var = 0.f;
#pragma unroll
  for (int e = 0; e < 16; ++e) { float d0 = acc[e] - mean; var += d0 * d0; }
  var *= 0.0625f;
  float rstd = rsqrtf(var + 1e-5f);

  // entmax 1.5: z = v/2 - max
  float z[16];
#pragma unroll
  for (int e = 0; e < 16; ++e) z[e] = (acc[e] - mean) * rstd * 0.5f;
  float zm = z[0];
#pragma unroll
  for (int e = 1; e < 16; ++e) zm = fmaxf(zm, z[e]);
#pragma unroll
  for (int e = 0; e < 16; ++e) z[e] -= zm;

  // rank-sort (descending) via LDS scatter
  int rk[16];
#pragma unroll
  for (int e = 0; e < 16; ++e) {
    int c = 0;
#pragma unroll
    for (int j = 0; j < 16; ++j)
      c += (z[j] > z[e]) || ((z[j] == z[e]) && (j < e));
    rk[e] = c;
  }
  if (lane == 0) {
#pragma unroll
    for (int e = 0; e < 16; ++e) zsort[wave][rk[e]] = z[e];
  }
  __syncthreads();
  float zs[16];
#pragma unroll
  for (int k = 0; k < 16; ++k) zs[k] = zsort[wave][k];

  // tau* via cumulative stats; support is a prefix -> last k with (tau_k <= zs_k)
  float cs = 0.f, css = 0.f, tau_star = 0.f;
#pragma unroll
  for (int k = 0; k < 16; ++k) {
    cs += zs[k]; css += zs[k] * zs[k];
    float kk = (float)(k + 1);
    float mn = cs / kk, msq = css / kk;
    float ss = kk * (msq - mn * mn);
    float delta = (1.f - ss) / kk;
    float tk = mn - sqrtf(fmaxf(delta, 0.f));
    if (tk <= zs[k]) tau_star = tk;
  }
  float a[16];
#pragma unroll
  for (int e = 0; e < 16; ++e) {
    float q = fmaxf(z[e] - tau_star, 0.f);
    a[e] = q * q;
  }
  // g1 = a @ C1, g2 = a @ C2  (C1/C2: [16,512], L2-resident)
  for (int rr = lane; rr < 512; rr += 32) {
    float s1 = 0.f, s2 = 0.f;
#pragma unroll
    for (int e = 0; e < 16; ++e) {
      s1 += a[e] * C1[e * 512 + rr];
      s2 += a[e] * C2[e * 512 + rr];
    }
    g1[(size_t)t * 512 + rr] = s1;
    g2[(size_t)t * 512 + rr] = s2;
  }
}

// ---------- bf16 WMMA GEMM: C[M,N] = A[M,K] * B[N,K]^T, fused epilogue ----------
// MODE 0: out_bf16 = (acc + bias[n]) * gate[m,n]
// MODE 1: out_bf16 = gelu_exact(acc)
// MODE 2: out_f32  = acc
#define BM 128
#define BN 128
#define BK 32
#define LDT 40  // padded halves/row: 80 B (16B-aligned, conflict-friendly)

template <int MODE>
__global__ __launch_bounds__(256) void gemm_bf16_kernel(
    const unsigned short* __restrict__ A,   // [M,K] bf16 row-major
    const unsigned short* __restrict__ B,   // [N,K] bf16 row-major (k contiguous)
    void* __restrict__ Out,
    const float* __restrict__ bias,         // [N] or null
    const float* __restrict__ gate,         // [M,N] or null
    int M, int N, int K) {
  __shared__ unsigned short As[2][BM * LDT];
  __shared__ unsigned short Bs[2][BN * LDT];

  const int tid = threadIdx.x;
  const int m0 = blockIdx.y * BM, n0 = blockIdx.x * BN;
  const int wave = tid >> 5, lane = tid & 31;
  const int wrow = wave >> 2, wcol = wave & 3;  // 2 x 4 wave grid -> 64x32 per wave
  const int l16 = lane & 15, lhi = lane >> 4;

  // global->LDS staging: thread t loads 16 halves of row (t>>1), segment (t&1)
  const int lr = tid >> 1, seg = tid & 1;
  const unsigned short* Ag = A + (size_t)(m0 + lr) * K + seg * 16;
  const unsigned short* Bg = B + (size_t)(n0 + lr) * K + seg * 16;
  const int sOff = lr * LDT + seg * 16;

  v8f acc[4][2] = {};

  auto compute = [&](int buf) {
    Frag af[4], bfr[2];
    const int c0 = lhi * 8;        // A frag: halves [c0,c0+8) and [c0+16,c0+24)
    const int k0 = lhi * 8;        // B frag: halves [k0,k0+8) and [k0+16,k0+24)
#pragma unroll
    for (int mt = 0; mt < 4; ++mt) {
      const unsigned short* base = &As[buf][(wrow * 64 + mt * 16 + l16) * LDT];
      af[mt].u[0] = *(const uint4*)(base + c0);
      af[mt].u[1] = *(const uint4*)(base + c0 + 16);
    }
#pragma unroll
    for (int nt = 0; nt < 2; ++nt) {
      const unsigned short* base = &Bs[buf][(wcol * 32 + nt * 16 + l16) * LDT];
      bfr[nt].u[0] = *(const uint4*)(base + k0);
      bfr[nt].u[1] = *(const uint4*)(base + k0 + 16);
    }
#pragma unroll
    for (int mt = 0; mt < 4; ++mt)
#pragma unroll
      for (int nt = 0; nt < 2; ++nt)
        acc[mt][nt] = __builtin_amdgcn_wmma_f32_16x16x32_bf16(
            false, af[mt].v, false, bfr[nt].v, (short)0, acc[mt][nt],
            false, false);
  };

  const int nk = K / BK;
  int buf = 0;

#if USE_ASYNC_LDS
  // Direct global -> LDS via GLOBAL_LOAD_ASYNC_TO_LDS_B128 (ASYNCcnt tracked).
  auto issue = [&](int kt, int tb) {
    const unsigned short* ga = Ag + (size_t)kt * BK;
    const unsigned short* gb = Bg + (size_t)kt * BK;
    unsigned short* la = &As[tb][sOff];
    unsigned short* lb = &Bs[tb][sOff];
    async_b128(ga, la);
    async_b128(ga + 8, la + 8);
    async_b128(gb, lb);
    async_b128(gb + 8, lb + 8);
  };
  issue(0, 0);
  __builtin_amdgcn_s_wait_asynccnt(0);
  __syncthreads();
  for (int kt = 0; kt < nk; ++kt) {
    const bool more = (kt + 1 < nk);
    // Safe: barrier at end of previous iteration guarantees all waves finished
    // reading buf^1 before new data lands in it.
    if (more) issue(kt + 1, buf ^ 1);
    if (kt + 2 < nk) {
      __builtin_prefetch(Ag + (size_t)(kt + 2) * BK, 0, 1);
      __builtin_prefetch(Bg + (size_t)(kt + 2) * BK, 0, 1);
    }
    compute(buf);
    if (more) {
      __builtin_amdgcn_s_wait_asynccnt(0);
      __syncthreads();
      buf ^= 1;
    }
  }
#else
  // Fallback: global -> VGPR -> LDS double buffering.
  uint4 ra0, ra1, rb0, rb1;
  auto gload = [&](int kt) {
    const uint4* pa = (const uint4*)(Ag + (size_t)kt * BK);
    ra0 = pa[0]; ra1 = pa[1];
    const uint4* pb = (const uint4*)(Bg + (size_t)kt * BK);
    rb0 = pb[0]; rb1 = pb[1];
  };
  auto lstore = [&](int tb) {
    uint4* sa = (uint4*)&As[tb][sOff];
    sa[0] = ra0; sa[1] = ra1;
    uint4* sb = (uint4*)&Bs[tb][sOff];
    sb[0] = rb0; sb[1] = rb1;
  };
  gload(0);
  lstore(0);
  __syncthreads();
  for (int kt = 0; kt < nk; ++kt) {
    const bool more = (kt + 1 < nk);
    if (more) gload(kt + 1);
    if (kt + 2 < nk) {
      __builtin_prefetch(Ag + (size_t)(kt + 2) * BK, 0, 1);
      __builtin_prefetch(Bg + (size_t)(kt + 2) * BK, 0, 1);
    }
    compute(buf);
    if (more) {
      __syncthreads();
      lstore(buf ^ 1);
      __syncthreads();
      buf ^= 1;
    }
  }
#endif

  // epilogue: lane L, vgpr v -> m = base + lhi*8 + v, n = base + l16
#pragma unroll
  for (int mt = 0; mt < 4; ++mt) {
#pragma unroll
    for (int nt = 0; nt < 2; ++nt) {
      const int n = n0 + wcol * 32 + nt * 16 + l16;
      const int mb = m0 + wrow * 64 + mt * 16 + lhi * 8;
      float bia = (MODE == 0) ? bias[n] : 0.f;
#pragma unroll
      for (int v = 0; v < 8; ++v) {
        const int m = mb + v;
        float val = acc[mt][nt][v];
        if (MODE == 0) {
          val = (val + bia) * gate[(size_t)m * N + n];
          ((unsigned short*)Out)[(size_t)m * N + n] = f2bf(val);
        } else if (MODE == 1) {
          ((unsigned short*)Out)[(size_t)m * N + n] = f2bf(gelu_exact(val));
        } else {
          ((float*)Out)[(size_t)m * N + n] = val;
        }
      }
    }
  }
}

// ---------- launch ----------
extern "C" void kernel_launch(void* const* d_in, const int* in_sizes, int n_in,
                              void* d_out, int out_size, void* d_ws,
                              size_t ws_size, hipStream_t stream) {
  (void)in_sizes; (void)n_in; (void)out_size; (void)ws_size;
  const float* x     = (const float*)d_in[0];   // [16384,1024]
  const float* proj  = (const float*)d_in[1];   // [1024,16]
  const float* C1    = (const float*)d_in[2];   // [16,512]
  const float* Win1  = (const float*)d_in[3];   // [1025,512]
  const float* Wout1 = (const float*)d_in[4];   // [4096,512]
  const float* C2    = (const float*)d_in[5];   // [16,512]
  const float* Win2  = (const float*)d_in[6];   // [4097,512]
  const float* Wout2 = (const float*)d_in[7];   // [1024,512]

  const int NT = 16384, D = 1024, O = 4096, R = 512, D2 = 1024;

  char* w = (char*)d_ws;
  auto carve = [&](size_t bytes) {
    char* p = w;
    w += (bytes + 255) & ~(size_t)255;
    return p;
  };
  unsigned short* xb16 = (unsigned short*)carve((size_t)NT * D * 2);
  unsigned short* w1T  = (unsigned short*)carve((size_t)R * D * 2);
  unsigned short* w2T  = (unsigned short*)carve((size_t)R * O * 2);
  unsigned short* wo1b = (unsigned short*)carve((size_t)O * R * 2);
  unsigned short* wo2b = (unsigned short*)carve((size_t)D2 * R * 2);
  float* g1 = (float*)carve((size_t)NT * R * 4);
  float* g2 = (float*)carve((size_t)NT * R * 4);
  unsigned short* z1 = (unsigned short*)carve((size_t)NT * R * 2);
  unsigned short* h16 = (unsigned short*)carve((size_t)NT * O * 2);
  unsigned short* z2 = (unsigned short*)carve((size_t)NT * R * 2);

  const float* bias1 = Win1 + (size_t)D * R;  // row D of [D+1,R]
  const float* bias2 = Win2 + (size_t)O * R;  // row O of [O+1,R]

  // converts
  {
    size_t n8 = (size_t)NT * D / 8;
    cvt_bf16_kernel<<<(unsigned)((n8 + 255) / 256), 256, 0, stream>>>(x, xb16, n8);
    n8 = (size_t)O * R / 8;
    cvt_bf16_kernel<<<(unsigned)((n8 + 255) / 256), 256, 0, stream>>>(Wout1, wo1b, n8);
    n8 = (size_t)D2 * R / 8;
    cvt_bf16_kernel<<<(unsigned)((n8 + 255) / 256), 256, 0, stream>>>(Wout2, wo2b, n8);
    size_t tot = (size_t)R * D;
    cvt_T_kernel<<<(unsigned)((tot + 255) / 256), 256, 0, stream>>>(Win1, w1T, D, tot);
    tot = (size_t)R * O;
    cvt_T_kernel<<<(unsigned)((tot + 255) / 256), 256, 0, stream>>>(Win2, w2T, O, tot);
  }

  // gating (fp32): a, then g1 = a@C1, g2 = a@C2
  gate_kernel<<<NT / 8, 256, 0, stream>>>(x, proj, C1, C2, g1, g2);

  // layer 1: z1 = (x@Win1 + b1) * g1        [16384,512], K=1024
  gemm_bf16_kernel<0><<<dim3(R / BN, NT / BM), 256, 0, stream>>>(
      xb16, w1T, (void*)z1, bias1, g1, NT, R, D);
  // h = gelu(z1 @ Wout1^T)                  [16384,4096], K=512
  gemm_bf16_kernel<1><<<dim3(O / BN, NT / BM), 256, 0, stream>>>(
      z1, wo1b, (void*)h16, nullptr, nullptr, NT, O, R);
  // layer 2: z2 = (h@Win2 + b2) * g2        [16384,512], K=4096
  gemm_bf16_kernel<0><<<dim3(R / BN, NT / BM), 256, 0, stream>>>(
      h16, w2T, (void*)z2, bias2, g2, NT, R, O);
  // out = z2 @ Wout2^T (f32)                [16384,1024], K=512
  gemm_bf16_kernel<2><<<dim3(D2 / BN, NT / BM), 256, 0, stream>>>(
      z2, wo2b, d_out, nullptr, nullptr, NT, D2, R);
}
